// AssociativeAttention_62397284876988
// MI455X (gfx1250) — matrix operand order repaired and checked by reading, not compiled, not run on hardware
//
#include <hip/hip_runtime.h>
#include <math.h>

// ---------------- problem constants ----------------
#define LQ   1024
#define DM   768
#define NH   12
#define HD   64
#define EPSV 1e-5f
#define HLh  (NH * LQ * HD)   // 786432
#define HL_  (NH * LQ)        // 12288

typedef __attribute__((ext_vector_type(16))) __bf16 v16bf;
typedef __attribute__((ext_vector_type(8)))  __bf16 v8bf;
typedef __attribute__((ext_vector_type(8)))  float  v8f;

#define V8Z {0.f,0.f,0.f,0.f,0.f,0.f,0.f,0.f}

// ------------- WMMA helpers (CDNA5 16x16x32 bf16, fp32 acc) -------------
__device__ __forceinline__ v8f wmma_bf(v16bf a, v16bf b, v8f c) {
  return __builtin_amdgcn_wmma_f32_16x16x32_bf16(false, a, false, b, (short)0, c,
                                                 false, false);
}

// A tile 16x32 (row-major, ld elems). ISA layout: lanes0-15 rows, lane-half hs
// selects K {0-7,16-23} vs {8-15,24-31}. Two aligned 16B loads per lane.
// REQUIRES: (m0+m)*ld + k0 + hs is a multiple of 8 elements (16B).
__device__ __forceinline__ v16bf ld_a(const __bf16* p0, int ld, int m0, int k0) {
  int lane = threadIdx.x & 31;
  int m = lane & 15, hs = (lane >> 4) << 3;
  const __bf16* p = p0 + (size_t)(m0 + m) * ld + k0 + hs;
  v8bf lo = *(const v8bf*)(p);
  v8bf hi = *(const v8bf*)(p + 16);
  v16bf r;
#pragma unroll
  for (int e = 0; e < 8; ++e) { r[e] = lo[e]; r[8 + e] = hi[e]; }
  return r;
}

// B tile 32x16 (row-major [K][N]). Lane = N column; K {0-7,16-23} vs {8-15,24-31} per lane-half (strided loads).
__device__ __forceinline__ v16bf ld_b(const __bf16* p0, int ld, int k0, int n0) {
  int lane = threadIdx.x & 31;
  int n = lane & 15, hs = (lane >> 4) << 3;
  const __bf16* p = p0 + (size_t)(k0 + hs) * ld + n0 + n;
  v16bf r;
#pragma unroll
  for (int e = 0; e < 8; ++e) { r[e] = p[(size_t)e * ld]; r[8 + e] = p[(size_t)(16 + e) * ld]; }
  return r;
}

__device__ __forceinline__ float wave_sum(float v) {
#pragma unroll
  for (int off = 16; off > 0; off >>= 1) v += __shfl_xor(v, off, 32);
  return v;
}

// ---------------- kernel 1: weight/filter bf16 prep ----------------
__global__ void __launch_bounds__(256) prep_kernel(
    const float* wq, const float* wk, const float* wv, const float* wo,
    const float* filters, __bf16* w3, __bf16* wob, __bf16* ftb) {
  const int NW3 = DM * 3 * DM, NWO = DM * DM, NFT = HD * LQ;
  int idx = blockIdx.x * 256 + threadIdx.x;
  if (idx < NW3) {                       // W3[k][which*768+col], K-major
    int k = idx / (3 * DM), j = idx % (3 * DM);
    int which = j / DM, col = j - which * DM;
    const float* src = which == 0 ? wq : which == 1 ? wk : wv;
    w3[idx] = (__bf16)src[k * DM + col];
  } else if (idx < NW3 + NWO) {
    int i = idx - NW3;
    wob[i] = (__bf16)wo[i];
  } else if (idx < NW3 + NWO + NFT) {    // filters transposed [c][l]
    int i = idx - NW3 - NWO;
    int c = i >> 10, l = i & 1023;
    ftb[i] = (__bf16)filters[l * HD + c];
  }
}

// ---------------- kernel 2: causal conv as Toeplitz WMMA GEMM ----------------
// Per channel c: y[l, head] = sum_{s<=l} f[l-s, c] * x[s, head*64+c].
// A built branch-free from 16 row-shifted reversed filter copies:
//   frevs[m][u] = f[1024+m-u]  =>  A(m0,k0) = ld_a(frevs, 1056, 0, 1024+k0-m0)
// x-panel staged in two 512-row passes; persistent per-wave accumulators.
// Inner K-loop software pipelined + unrolled x2 (ping-pong register sets).
__global__ void __launch_bounds__(256) conv_kernel(const float* x, const __bf16* ft,
                                                   __bf16* xt) {
  __shared__ __bf16 fcol[LQ];
  __shared__ __bf16 frevs[16 * 1056];
  __shared__ __bf16 xcol[512 * 16];
  int c = blockIdx.x;
  int tid = threadIdx.x, lane = tid & 31;
  int wave = __builtin_amdgcn_readfirstlane(tid >> 5);   // wave-uniform SGPR
  int nn = lane & 15, mb = (lane >> 4) << 3;

  for (int i = tid; i < LQ; i += 256) fcol[i] = ft[c * LQ + i];
  __syncthreads();
  for (int i = tid; i < 16 * 1056; i += 256) {
    int m = i / 1056, u = i - m * 1056;
    int idx = 1024 + m - u;
    frevs[i] = (idx >= 0 && idx < LQ) ? fcol[idx] : (__bf16)0.f;
  }

  v8f acc[8];
#pragma unroll
  for (int t = 0; t < 8; ++t) acc[t] = V8Z;

  for (int pass = 0; pass < 2; ++pass) {
    int kbase = pass * 512;
    __syncthreads();
    for (int i = tid; i < 512 * 16; i += 256) {
      int sl = i >> 4, j = i & 15;
      float v = (j < NH) ? x[(size_t)(kbase + sl) * DM + j * HD + c] : 0.f;
      xcol[i] = (__bf16)v;
    }
    __syncthreads();
#pragma unroll
    for (int t = 0; t < 8; ++t) {
      int m0 = (wave + 8 * t) * 16;
      int kend = m0 + 16 < kbase + 512 ? m0 + 16 : kbase + 512;
      int k0 = kbase;
      if (k0 < kend) {
        v16bf a = ld_a(frevs, 1056, 0, 1024 + k0 - m0);
        v16bf b = ld_b(xcol, 16, k0 - kbase, 0);
#pragma unroll 2
        for (k0 += 32; k0 < kend; k0 += 32) {
          v16bf an = ld_a(frevs, 1056, 0, 1024 + k0 - m0);
          v16bf bn = ld_b(xcol, 16, k0 - kbase, 0);
          acc[t] = wmma_bf(a, b, acc[t]);
          a = an; b = bn;
        }
        acc[t] = wmma_bf(a, b, acc[t]);
      }
    }
  }
  if (nn < NH) {
#pragma unroll
    for (int t = 0; t < 8; ++t) {
      int m0 = (wave + 8 * t) * 16;
#pragma unroll
      for (int r = 0; r < 8; ++r)
        xt[(size_t)(m0 + mb + r) * DM + nn * HD + c] = (__bf16)acc[t][r];
    }
  }
}

// ---------------- kernel 3: fused QKV projection GEMM ----------------
__global__ void __launch_bounds__(256) qkv_gemm(
    const __bf16* xt, const __bf16* w3, const float* qb, const float* kb,
    const float* vb, float* qraw, float* kraw, float* vraw) {
  int lane = threadIdx.x & 31;
  int wave = __builtin_amdgcn_readfirstlane(threadIdx.x >> 5);
  int n0 = blockIdx.x * 256 + wave * 32;
  int m0 = blockIdx.y * 32;
  v8f a00 = V8Z, a01 = V8Z, a10 = V8Z, a11 = V8Z;
  // software-pipelined K loop (unroll x2: ping-pong register sets)
  v16bf A0 = ld_a(xt, DM, m0, 0);
  v16bf A1 = ld_a(xt, DM, m0 + 16, 0);
  v16bf B0 = ld_b(w3, 3 * DM, 0, n0);
  v16bf B1 = ld_b(w3, 3 * DM, 0, n0 + 16);
#pragma unroll 2
  for (int k0 = 32; k0 < DM; k0 += 32) {
    __builtin_prefetch((const void*)(w3 + (size_t)(k0 + 32 + lane) * (3 * DM) + n0), 0, 0);
    v16bf A0n = ld_a(xt, DM, m0, k0);
    v16bf A1n = ld_a(xt, DM, m0 + 16, k0);
    v16bf B0n = ld_b(w3, 3 * DM, k0, n0);
    v16bf B1n = ld_b(w3, 3 * DM, k0, n0 + 16);
    a00 = wmma_bf(A0, B0, a00);
    a01 = wmma_bf(A0, B1, a01);
    a10 = wmma_bf(A1, B0, a10);
    a11 = wmma_bf(A1, B1, a11);
    A0 = A0n; A1 = A1n; B0 = B0n; B1 = B1n;
  }
  a00 = wmma_bf(A0, B0, a00);
  a01 = wmma_bf(A0, B1, a01);
  a10 = wmma_bf(A1, B0, a10);
  a11 = wmma_bf(A1, B1, a11);

  int nn = lane & 15, mb = (lane >> 4) << 3;
#pragma unroll
  for (int t = 0; t < 4; ++t) {
    v8f acc = t == 0 ? a00 : t == 1 ? a01 : t == 2 ? a10 : a11;
    int j = n0 + ((t & 1) ? 16 : 0) + nn;
    int mrow = m0 + ((t >> 1) ? 16 : 0) + mb;
    int which = j / DM, d = j - which * DM;
    int head = d >> 6, cc = d & 63;
    const float* bias = which == 0 ? qb : which == 1 ? kb : vb;
    float* dst = which == 0 ? qraw : which == 1 ? kraw : vraw;
    float bv = bias[d];
#pragma unroll
    for (int r = 0; r < 8; ++r)
      dst[(size_t)head * (LQ * HD) + (size_t)(mrow + r) * HD + cc] = acc[r] + bv;
  }
}

// ---------------- kernel 4: l2norm + sim + collapsed gate ----------------
// gate_logits = kv_scale * v^T (Wg k) + wg_b ; gates = leaky(gl)^2 + eps.
__global__ void __launch_bounds__(64) norm_kernel(
    const float* qraw, const float* kraw, const float* vraw,
    __bf16* qn, __bf16* kn, __bf16* vn, float* vnf,
    float* sim, float* gates, const float* wg_w, const float* wg_b,
    const float* qk_scale, const float* kv_scale) {
  int blk = blockIdx.x;              // head*1024 + l
  int head = blk >> 10;
  int p = threadIdx.x;
  size_t base = (size_t)blk * HD;
  __shared__ float part[2];
  __shared__ float shk[64], shv[64];
  float qv = qraw[base + p], kv = kraw[base + p], vv = vraw[base + p];
  auto bsum = [&](float v) -> float {
    float s = wave_sum(v);
    __syncthreads();
    if ((p & 31) == 0) part[p >> 5] = s;
    __syncthreads();
    return part[0] + part[1];
  };
  float qr = qv / fmaxf(sqrtf(bsum(qv * qv)), 1e-12f);
  float kr = kv / fmaxf(sqrtf(bsum(kv * kv)), 1e-12f);
  float vr = vv / fmaxf(sqrtf(bsum(vv * vv)), 1e-12f);
  qn[base + p] = (__bf16)qr;
  kn[base + p] = (__bf16)kr;
  vn[base + p] = (__bf16)vr;
  vnf[base + p] = vr;
  shk[p] = kr; shv[p] = vr;
  float s = bsum(qr * kr) * qk_scale[head];
  if (p == 0) sim[blk] = s;
  __syncthreads();
  float glp = 0.f;
#pragma unroll 8
  for (int n = 0; n < HD; ++n) glp += wg_w[p * HD + n] * shk[n];
  float gl = bsum(shv[p] * glp) * kv_scale[head] + wg_b[0];
  float lk = gl > 0.f ? gl : 0.01f * gl;
  if (p == 0) gates[blk] = lk * lk + EPSV;
}

// ---------------- kernel 5: (m,s,n) softmax scan + gate prefix sum ----------------
__global__ void __launch_bounds__(64) scan_kernel(
    const float* sim, const float* gates, const float* vnf,
    float* la, float* sw, float* ginv) {
  int head = blockIdx.x, d = threadIdx.x;
  size_t vb = (size_t)head * (LQ * HD);
  float m = -INFINITY, s = 0.f, n = 0.f, G = 0.f;
  for (int l = 0; l < LQ; ++l) {
    float sm = sim[head * LQ + l];
    float g  = gates[head * LQ + l];
    float v  = vnf[vb + (size_t)l * HD + d];
    float mn = fmaxf(m, sm);
    float eo = __expf(m - mn);
    float pe = __expf(sm - mn);
    s = s * eo + pe;
    n = n * eo + pe * v;
    G += g;
    m = mn;
    la[vb + (size_t)l * HD + d] = n / (s + EPSV);
    if (d == 0) {
      sw[head * LQ + l]   = pe / (s + EPSV);
      ginv[head * LQ + l] = 1.f / (G + EPSV);
    }
  }
}

// ---------------- kernel 6: chunked causal linear attention + lerp ----------------
// ctx[t] = kvsc * [ (mask . (Q V^T) diag(g)) K + Q S ] * Ginv; S += (g.V)^T K.
__global__ void __launch_bounds__(512) attn_kernel(
    const __bf16* qn, const __bf16* kn, const __bf16* vn,
    const float* gates, const float* ginv, const float* sw, const float* la,
    const float* kv_scale, __bf16* ybf) {
  __shared__ __bf16 Qc[64 * 64], Kc[64 * 64], Vt[64 * 64], GvT[64 * 64];
  __shared__ __bf16 Wm[64 * 64], Sl[64 * 64];
  int head = blockIdx.x;
  int tid = threadIdx.x, lane = tid & 31;
  int wave = __builtin_amdgcn_readfirstlane(tid >> 5);
  int ti = wave >> 2, tj = wave & 3;
  int nn = lane & 15, mb = (lane >> 4) << 3;
  size_t hb = (size_t)head * (LQ * HD);
  float kvsc = kv_scale[head];
  v8f S = V8Z;
  for (int i = tid; i < 4096; i += 512) Sl[i] = (__bf16)0.f;
  __syncthreads();

  for (int ch = 0; ch < 16; ++ch) {
    int l0 = ch * 64;
    {   // staging: 512 threads, one 8-wide chunk each (vectorized global side)
      int r = tid >> 3, c8 = (tid & 7) << 3;
      size_t src = hb + (size_t)(l0 + r) * HD + c8;
      *(v8bf*)(Qc + r * 64 + c8) = *(const v8bf*)(qn + src);
      *(v8bf*)(Kc + r * 64 + c8) = *(const v8bf*)(kn + src);
      v8bf vv = *(const v8bf*)(vn + src);
      float g = gates[head * LQ + l0 + r];
#pragma unroll
      for (int e = 0; e < 8; ++e) {
        Vt[(c8 + e) * 64 + r]  = vv[e];
        GvT[(c8 + e) * 64 + r] = (__bf16)(g * (float)vv[e]);
      }
    }
    __syncthreads();

    // A = Q V^T (K-dim = p); both K-step operand pairs issued up front
    v8f acc = V8Z;
    {
      v16bf a0 = ld_a(Qc, 64, ti * 16, 0),  b0 = ld_b(Vt, 64, 0, tj * 16);
      v16bf a1 = ld_a(Qc, 64, ti * 16, 32), b1 = ld_b(Vt, 64, 32, tj * 16);
      acc = wmma_bf(a0, b0, acc);
      acc = wmma_bf(a1, b1, acc);
    }

    // causal mask + per-source gate, write bf16 W
    {
      int sl = tj * 16 + nn;
      float gsv = gates[head * LQ + l0 + sl];
#pragma unroll
      for (int r = 0; r < 8; ++r) {
        int tl = ti * 16 + mb + r;
        float w = (tl >= sl) ? acc[r] * gsv : 0.f;
        Wm[tl * 64 + sl] = (__bf16)w;
      }
    }
    __syncthreads();

    // ctx = W K + Q S_prev
    v8f c2 = V8Z;
    {
      v16bf a0 = ld_a(Wm, 64, ti * 16, 0),  b0 = ld_b(Kc, 64, 0, tj * 16);
      v16bf a1 = ld_a(Wm, 64, ti * 16, 32), b1 = ld_b(Kc, 64, 32, tj * 16);
      v16bf a2 = ld_a(Qc, 64, ti * 16, 0),  b2 = ld_b(Sl, 64, 0, tj * 16);
      v16bf a3 = ld_a(Qc, 64, ti * 16, 32), b3 = ld_b(Sl, 64, 32, tj * 16);
      c2 = wmma_bf(a0, b0, c2);
      c2 = wmma_bf(a1, b1, c2);
      c2 = wmma_bf(a2, b2, c2);
      c2 = wmma_bf(a3, b3, c2);
    }
    {
      int d = tj * 16 + nn;
#pragma unroll
      for (int r = 0; r < 8; ++r) {
        int tl = ti * 16 + mb + r;
        int lg = l0 + tl;
        float ctx = kvsc * c2[r] * ginv[head * LQ + lg];
        float lav = la[hb + (size_t)lg * HD + d];
        float swv = sw[head * LQ + lg];
        float o = ctx + (lav - ctx) * swv;
        ybf[(size_t)lg * DM + head * HD + d] = (__bf16)o;
      }
    }
    __syncthreads();   // all reads of Sl done before update

    // S += (g.V)^T K
    {
      v16bf a0 = ld_a(GvT, 64, ti * 16, 0),  b0 = ld_b(Kc, 64, 0, tj * 16);
      v16bf a1 = ld_a(GvT, 64, ti * 16, 32), b1 = ld_b(Kc, 64, 32, tj * 16);
      S = wmma_bf(a0, b0, S);
      S = wmma_bf(a1, b1, S);
    }
#pragma unroll
    for (int r = 0; r < 8; ++r)
      Sl[(ti * 16 + mb + r) * 64 + tj * 16 + nn] = (__bf16)S[r];
    __syncthreads();
  }
}

// ---------------- kernel 7: output projection GEMM ----------------
__global__ void __launch_bounds__(256) out_gemm(
    const __bf16* ybf, const __bf16* wob, const float* ob, float* out) {
  int lane = threadIdx.x & 31;
  int wave = __builtin_amdgcn_readfirstlane(threadIdx.x >> 5);
  int n0 = blockIdx.x * 256 + wave * 32;
  int m0 = blockIdx.y * 32;
  v8f a00 = V8Z, a01 = V8Z, a10 = V8Z, a11 = V8Z;
  v16bf A0 = ld_a(ybf, DM, m0, 0);
  v16bf A1 = ld_a(ybf, DM, m0 + 16, 0);
  v16bf B0 = ld_b(wob, DM, 0, n0);
  v16bf B1 = ld_b(wob, DM, 0, n0 + 16);
#pragma unroll 2
  for (int k0 = 32; k0 < DM; k0 += 32) {
    __builtin_prefetch((const void*)(wob + (size_t)(k0 + 32 + lane) * DM + n0), 0, 0);
    v16bf A0n = ld_a(ybf, DM, m0, k0);
    v16bf A1n = ld_a(ybf, DM, m0 + 16, k0);
    v16bf B0n = ld_b(wob, DM, k0, n0);
    v16bf B1n = ld_b(wob, DM, k0, n0 + 16);
    a00 = wmma_bf(A0, B0, a00);
    a01 = wmma_bf(A0, B1, a01);
    a10 = wmma_bf(A1, B0, a10);
    a11 = wmma_bf(A1, B1, a11);
    A0 = A0n; A1 = A1n; B0 = B0n; B1 = B1n;
  }
  a00 = wmma_bf(A0, B0, a00);
  a01 = wmma_bf(A0, B1, a01);
  a10 = wmma_bf(A1, B0, a10);
  a11 = wmma_bf(A1, B1, a11);

  int nn = lane & 15, mb = (lane >> 4) << 3;
#pragma unroll
  for (int t = 0; t < 4; ++t) {
    v8f acc = t == 0 ? a00 : t == 1 ? a01 : t == 2 ? a10 : a11;
    int j = n0 + ((t & 1) ? 16 : 0) + nn;
    int mrow = m0 + ((t >> 1) ? 16 : 0) + mb;
    float bv = ob[j];
#pragma unroll
    for (int r = 0; r < 8; ++r)
      out[(size_t)(mrow + r) * DM + j] = acc[r] + bv;
  }
}

// ---------------- launcher ----------------
extern "C" void kernel_launch(void* const* d_in, const int* in_sizes, int n_in,
                              void* d_out, int out_size, void* d_ws, size_t ws_size,
                              hipStream_t stream) {
  const float* x        = (const float*)d_in[0];
  const float* filters  = (const float*)d_in[1];
  const float* wq_w     = (const float*)d_in[2];
  const float* wq_b     = (const float*)d_in[3];
  const float* wk_w     = (const float*)d_in[4];
  const float* wk_b     = (const float*)d_in[5];
  const float* wv_w     = (const float*)d_in[6];
  const float* wv_b     = (const float*)d_in[7];
  const float* wo_w     = (const float*)d_in[8];
  const float* wo_b     = (const float*)d_in[9];
  const float* wg_w     = (const float*)d_in[10];
  const float* wg_b     = (const float*)d_in[11];
  const float* qk_scale = (const float*)d_in[12];
  const float* kv_scale = (const float*)d_in[13];

  char* w = (char*)d_ws;
  auto alloc = [&](size_t bytes) -> char* {
    char* p = w;
    w += (bytes + 255) & ~(size_t)255;
    return p;
  };
  __bf16* xt   = (__bf16*)alloc((size_t)LQ * DM * 2);
  __bf16* w3   = (__bf16*)alloc((size_t)DM * 3 * DM * 2);
  __bf16* wob  = (__bf16*)alloc((size_t)DM * DM * 2);
  __bf16* ftb  = (__bf16*)alloc((size_t)HD * LQ * 2);
  float*  qraw = (float*)alloc((size_t)HLh * 4);
  float*  kraw = (float*)alloc((size_t)HLh * 4);
  float*  vraw = (float*)alloc((size_t)HLh * 4);
  __bf16* qn   = (__bf16*)alloc((size_t)HLh * 2);
  __bf16* kn   = (__bf16*)alloc((size_t)HLh * 2);
  __bf16* vn   = (__bf16*)alloc((size_t)HLh * 2);
  float*  vnf  = (float*)alloc((size_t)HLh * 4);
  float*  sim  = (float*)alloc((size_t)HL_ * 4);
  float*  gat  = (float*)alloc((size_t)HL_ * 4);
  float*  giv  = (float*)alloc((size_t)HL_ * 4);
  float*  swp  = (float*)alloc((size_t)HL_ * 4);
  float*  la   = (float*)alloc((size_t)HLh * 4);
  __bf16* ybf  = (__bf16*)alloc((size_t)LQ * DM * 2);

  const int PREP_N = DM * 3 * DM + DM * DM + HD * LQ;
  prep_kernel<<<(PREP_N + 255) / 256, 256, 0, stream>>>(wq_w, wk_w, wv_w, wo_w,
                                                        filters, w3, wob, ftb);
  conv_kernel<<<HD, 256, 0, stream>>>(x, ftb, xt);
  qkv_gemm<<<dim3(9, 32), 256, 0, stream>>>(xt, w3, wq_b, wk_b, wv_b,
                                            qraw, kraw, vraw);
  norm_kernel<<<HL_, 64, 0, stream>>>(qraw, kraw, vraw, qn, kn, vn, vnf,
                                      sim, gat, wg_w, wg_b, qk_scale, kv_scale);
  scan_kernel<<<NH, 64, 0, stream>>>(sim, gat, vnf, la, swp, giv);
  attn_kernel<<<NH, 512, 0, stream>>>(qn, kn, vn, gat, giv, swp, la,
                                      kv_scale, ybf);
  out_gemm<<<dim3(3, 32), 256, 0, stream>>>(ybf, wob, wo_b, (float*)d_out);
}
